// Mamba2ICL_89696097009922
// MI455X (gfx1250) — matrix operand order repaired and checked by reading, not compiled, not run on hardware
//
#include <hip/hip_runtime.h>
#include <hip/hip_bf16.h>

// ---- CDNA5 WMMA / TDM types ----
typedef __attribute__((ext_vector_type(16))) __bf16 v16bf;
typedef __attribute__((ext_vector_type(8)))  float  v8f;
typedef __attribute__((ext_vector_type(4)))  float  f32x4;
typedef __attribute__((ext_vector_type(4)))  unsigned int u32x4;
typedef __attribute__((ext_vector_type(4)))  int    i32x4;
typedef __attribute__((ext_vector_type(8)))  int    i32x8;

// ---- problem dims ----
#define D_MODEL   1024
#define D_STATE   64
#define D_CONV    4
#define D_INNER   2048
#define HEADDIM   64
#define NHEADS    32
#define CONV_DIM  (D_INNER + 2 * D_STATE)              // 2176
#define D_IN_PROJ (2 * D_INNER + 2 * D_STATE + NHEADS) // 4256
#define BATCH     8
#define SEQ       1024
#define ROWS      (BATCH * SEQ)                        // 8192

// GEMM tiling
#define BN        128   // block N tile
#define BK        32    // k-step
#define COL_STR   40    // LDS column stride in bf16 (80B: 64B data + 16B TDM pad)
#define NPAD_IN   (((D_IN_PROJ + BN - 1) / BN) * BN)   // 4352
#define NPAD_OUT  D_MODEL                              // 1024

__device__ inline __bf16 f2bf(float f) { return (__bf16)f; }  // native cvt
__device__ inline float silu(float x) { return x / (1.f + __expf(-x)); }

// =====================================================================
// One-time weight prep: W[K,N] f32 row-major -> Wt[Npad,K] bf16
// (transposed + padded with zero rows so GEMM edge tiles read zeros).
// =====================================================================
__global__ void __launch_bounds__(256) convert_transpose_kernel(
    const float* __restrict__ W, __bf16* __restrict__ Wt,
    int K, int N, int Npad)
{
    long idx = (long)blockIdx.x * 256 + threadIdx.x;   // over Npad*K
    if (idx >= (long)Npad * K) return;
    int k = (int)(idx % K);
    int n = (int)(idx / K);
    Wt[idx] = (n < N) ? f2bf(W[(size_t)k * N + n]) : (__bf16)0.f;
}

// =====================================================================
// GEMM: C[M,N] = A[M,K] x B[K,N]; A f32 row-major, B pre-transposed bf16
// (Bt[n][k], Npad rows). Block: 128 threads = 4 waves; tile 64(M)x128(N).
// Per k-step the 32x128 bf16 B slab is DMA'd into LDS by the Tensor Data
// Mover (wave 0 issues TENSOR_LOAD_TO_LDS; TDM pad gives 80B column stride,
// bank-conflict-free). A fragment f32 loads + cvt overlap the DMA; then
// s_wait_tensorcnt + barrier, and 8 WMMAs per wave off one A fragment.
// =====================================================================
__global__ void __launch_bounds__(128) wmma_gemm_bf16(
    const float* __restrict__ A, const __bf16* __restrict__ Bt,
    float* __restrict__ C, int M, int N, int K, int Npad)
{
    __shared__ __align__(32) __bf16 sBt[BN * COL_STR];  // [n][k] 64B + 16B pad

    const int tid  = threadIdx.x;
    const int wave = tid >> 5;
    const int lane = tid & 31;
    const int half = lane >> 4;   // 0: lanes 0-15, 1: lanes 16-31
    const int mr   = lane & 15;

    const int m0 = blockIdx.y * 64 + wave * 16;
    const int n0 = blockIdx.x * BN;

    v8f acc[8];
    for (int i = 0; i < 8; ++i)
        for (int j = 0; j < 8; ++j) acc[i][j] = 0.f;

    const float* arow = A + (size_t)(m0 + mr) * K;
    const unsigned lds_base = (unsigned)(size_t)&sBt[0];

    // D# group1 (constant across k): data_size=2B, pad_enable,
    // pad_interval code 3 (=16 DWORDs = one 64B tile row), pad_amount code 3
    // (=4 DWORDs = 16B) -> LDS row stride 80B. tile = 32(k) x 128(n),
    // tensor_dim0 = K, tensor_dim1 = Npad, dim0 stride = K elements.
    i32x8 g1;
    g1[0] = (1 << 16) | (1 << 20) | (3 << 22) | (3 << 25);
    g1[1] = (int)((unsigned)(K & 0xFFFF) << 16);                      // dim0[15:0]
    g1[2] = (int)(((unsigned)K >> 16) | ((unsigned)(Npad & 0xFFFF) << 16));
    g1[3] = (int)(((unsigned)Npad >> 16) | ((unsigned)BK << 16));     // tile_dim0
    g1[4] = BN;                                                       // tile_dim1
    g1[5] = K;                                                        // dim0_stride
    g1[6] = 0;
    g1[7] = 0;
    const i32x4 gz4 = {0, 0, 0, 0};
    const i32x8 gz8 = {0, 0, 0, 0, 0, 0, 0, 0};

    for (int k0 = 0; k0 < K; k0 += BK) {
        __syncthreads();   // previous iteration's fragment reads complete
        if (wave == 0) {
            unsigned long long ga =
                (unsigned long long)(size_t)(Bt + (size_t)n0 * K + k0);
            u32x4 g0;
            g0[0] = 1u;                       // count=1, user descriptor
            g0[1] = lds_base;                 // lds_addr
            g0[2] = (unsigned)ga;             // global_addr lo
            g0[3] = (unsigned)((ga >> 32) & 0x1FFFFFFu) | (2u << 30); // type=2
            __builtin_amdgcn_tensor_load_to_lds(g0, g1, gz4, gz4, gz8, 0);
        }

        // ---- A fragment (overlaps the TDM transfer) ----
        // A k order used here (contiguous, same as B): lane<16 K={0..15}, lane>=16 K={16..31}
        f32x4 a0 = *(const f32x4*)(arow + k0 + 16 * half);
        f32x4 a1 = *(const f32x4*)(arow + k0 + 16 * half + 4);
        f32x4 a2 = *(const f32x4*)(arow + k0 + 16 * half + 8);
        f32x4 a3 = *(const f32x4*)(arow + k0 + 16 * half + 12);
        v16bf af;
        for (int j = 0; j < 4; ++j) {
            af[j]      = f2bf(a0[j]);
            af[4 + j]  = f2bf(a1[j]);
            af[8 + j]  = f2bf(a2[j]);
            af[12 + j] = f2bf(a3[j]);
        }
        if (k0 + BK < K) __builtin_prefetch(arow + k0 + BK, 0, 1); // global_prefetch_b8

        if (wave == 0) __builtin_amdgcn_s_wait_tensorcnt(0);
        __syncthreads();   // B slab visible to all waves

        // ---- 8 WMMAs: B fragment = contiguous 16 bf16 (K = 16*half..+15) ----
        for (int nt = 0; nt < 8; ++nt) {
            const int n = nt * 16 + mr;
            v16bf bfrag = *(const v16bf*)(sBt + (size_t)n * COL_STR + 16 * half);
            acc[nt] = __builtin_amdgcn_wmma_f32_16x16x32_bf16(
                false, af, false, bfrag, (short)0, acc[nt], false, false);
        }
    }

    // ---- store: f32 D 16x16 layout: VGPR r -> M = r + 8*half, N = mr ----
    for (int nt = 0; nt < 8; ++nt) {
        const int n = n0 + nt * 16 + mr;
        if (n >= N) continue;
        for (int r = 0; r < 8; ++r) {
            int m = m0 + r + 8 * half;
            C[(size_t)m * N + n] = acc[nt][r];
        }
    }
}

// =====================================================================
// Depthwise causal conv (taps=4) + SiLU over the xBC slice of zxbcdt.
// =====================================================================
__global__ void __launch_bounds__(256) conv_silu_kernel(
    const float* __restrict__ zxbcdt, const float* __restrict__ conv_w,
    const float* __restrict__ conv_b, float* __restrict__ xBC)
{
    long idx = (long)blockIdx.x * 256 + threadIdx.x;
    if (idx >= (long)ROWS * CONV_DIM) return;
    int  c   = (int)(idx % CONV_DIM);
    long row = idx / CONV_DIM;
    int  t   = (int)(row % SEQ);
    float acc = conv_b[c];
    for (int k = 0; k < D_CONV; ++k) {
        int tt = t - (D_CONV - 1) + k;
        if (tt >= 0)
            acc += conv_w[k * CONV_DIM + c] *
                   zxbcdt[(size_t)(row + (tt - t)) * D_IN_PROJ + D_INNER + c];
    }
    xBC[(size_t)row * CONV_DIM + c] = silu(acc);
}

// =====================================================================
// dt = softplus(raw_dt + dt_bias)
// =====================================================================
__global__ void __launch_bounds__(256) dt_kernel(
    const float* __restrict__ zxbcdt, const float* __restrict__ dt_bias,
    float* __restrict__ dtb)
{
    long idx = (long)blockIdx.x * 256 + threadIdx.x;
    if (idx >= (long)ROWS * NHEADS) return;
    int  h   = (int)(idx % NHEADS);
    long row = idx / NHEADS;
    float v = zxbcdt[(size_t)row * D_IN_PROJ + D_INNER + CONV_DIM + h] + dt_bias[h];
    dtb[(size_t)row * NHEADS + h] = (v > 20.f) ? v : log1pf(__expf(v));
}

// =====================================================================
// Selective scan: one block per (batch, head). 256 threads, each lane group
// of 4 owns state row p (16 f32 states each). B/C/x staged in LDS per step;
// y reduced across the 4-lane group with wave32 shfl_xor.
// =====================================================================
__global__ void __launch_bounds__(256) scan_kernel(
    const float* __restrict__ xBC, const float* __restrict__ dtb,
    const float* __restrict__ A_log, const float* __restrict__ Dskip,
    float* __restrict__ y)
{
    __shared__ float sx[HEADDIM], sB[D_STATE], sC[D_STATE];
    const int tid  = threadIdx.x;
    const int head = blockIdx.x & (NHEADS - 1);
    const int b    = blockIdx.x >> 5;   // NHEADS == 32
    const float Ah = -__expf(A_log[head]);
    const float Dh = Dskip[head];
    const int p  = tid >> 2;            // 0..63 state row
    const int s0 = (tid & 3) << 4;      // 0,16,32,48 state col base

    float h[16];
    for (int j = 0; j < 16; ++j) h[j] = 0.f;

    for (int t = 0; t < SEQ; ++t) {
        const long row = (long)b * SEQ + t;
        const float* xr = xBC + (size_t)row * CONV_DIM;
        if (tid < 64)        sx[tid]        = xr[head * HEADDIM + tid];
        else if (tid < 128)  sB[tid - 64]   = xr[D_INNER + (tid - 64)];
        else if (tid < 192)  sC[tid - 128]  = xr[D_INNER + D_STATE + (tid - 128)];
        const float dtv = dtb[(size_t)row * NHEADS + head];
        __syncthreads();

        const float dA   = __expf(dtv * Ah);
        const float coef = dtv * sx[p];
        float ys = 0.f;
        for (int j = 0; j < 16; ++j) {
            h[j] = h[j] * dA + coef * sB[s0 + j];
            ys  += h[j] * sC[s0 + j];
        }
        ys += __shfl_xor(ys, 1, 32);
        ys += __shfl_xor(ys, 2, 32);
        if ((tid & 3) == 0)
            y[(size_t)row * D_INNER + head * HEADDIM + p] = ys + Dh * sx[p];
        __syncthreads();
    }
}

// =====================================================================
// y = y * silu(z); RMSNorm(y) * norm_w   (in-place on y)
// =====================================================================
__global__ void __launch_bounds__(256) gate_norm_kernel(
    float* __restrict__ y, const float* __restrict__ zxbcdt,
    const float* __restrict__ norm_w)
{
    __shared__ float red[8];
    __shared__ float s_scale;
    const long row = blockIdx.x;
    const int  tid = threadIdx.x;
    float g[8];
    float ss = 0.f;
    for (int j = 0; j < 8; ++j) {
        int c = tid + j * 256;
        float z = zxbcdt[(size_t)row * D_IN_PROJ + c];
        float v = y[(size_t)row * D_INNER + c] * silu(z);
        g[j] = v;
        ss += v * v;
    }
    for (int o = 16; o >= 1; o >>= 1) ss += __shfl_xor(ss, o, 32);
    if ((tid & 31) == 0) red[tid >> 5] = ss;
    __syncthreads();
    if (tid == 0) {
        float t2 = 0.f;
        for (int i = 0; i < 8; ++i) t2 += red[i];
        s_scale = rsqrtf(t2 / (float)D_INNER + 1e-5f);
    }
    __syncthreads();
    const float sc = s_scale;
    for (int j = 0; j < 8; ++j) {
        int c = tid + j * 256;
        y[(size_t)row * D_INNER + c] = g[j] * sc * norm_w[c];
    }
}

// =====================================================================
// final head GEMV: out[row] = dot(yi[row,:], head_w) + head_b
// =====================================================================
__global__ void __launch_bounds__(256) head_kernel(
    const float* __restrict__ yi, const float* __restrict__ head_w,
    const float* __restrict__ head_b, float* __restrict__ out)
{
    __shared__ float red[8];
    const long row = blockIdx.x;
    const int  tid = threadIdx.x;
    float acc = 0.f;
    for (int j = 0; j < 4; ++j) {
        int c = tid + j * 256;
        acc += yi[(size_t)row * D_MODEL + c] * head_w[c];
    }
    for (int o = 16; o >= 1; o >>= 1) acc += __shfl_xor(acc, o, 32);
    if ((tid & 31) == 0) red[tid >> 5] = acc;
    __syncthreads();
    if (tid == 0) {
        float t = 0.f;
        for (int i = 0; i < 8; ++i) t += red[i];
        out[row] = t + head_b[0];
    }
}

extern "C" void kernel_launch(void* const* d_in, const int* in_sizes, int n_in,
                              void* d_out, int out_size, void* d_ws, size_t ws_size,
                              hipStream_t stream)
{
    (void)in_sizes; (void)n_in; (void)out_size; (void)ws_size;
    const float* x       = (const float*)d_in[0];
    const float* W_in    = (const float*)d_in[1];
    const float* conv_w  = (const float*)d_in[2];
    const float* conv_b  = (const float*)d_in[3];
    const float* dt_bias = (const float*)d_in[4];
    const float* A_log   = (const float*)d_in[5];
    const float* Dskip   = (const float*)d_in[6];
    const float* norm_w  = (const float*)d_in[7];
    const float* W_out   = (const float*)d_in[8];
    const float* head_w  = (const float*)d_in[9];
    const float* head_b  = (const float*)d_in[10];

    float* ws     = (float*)d_ws;
    float* zxbcdt = ws;                                  // ROWS * D_IN_PROJ
    float* xBC    = zxbcdt + (size_t)ROWS * D_IN_PROJ;   // ROWS * CONV_DIM
    float* dtb    = xBC    + (size_t)ROWS * CONV_DIM;    // ROWS * NHEADS
    float* y      = dtb    + (size_t)ROWS * NHEADS;      // ROWS * D_INNER
    float* yi     = y      + (size_t)ROWS * D_INNER;     // ROWS * D_MODEL
    __bf16* wtin  = (__bf16*)(yi + (size_t)ROWS * D_MODEL); // NPAD_IN * D_MODEL
    __bf16* wtout = wtin + (size_t)NPAD_IN * D_MODEL;       // NPAD_OUT * D_INNER

    // 0) weight prep: bf16 + transpose (+zero-pad rows)
    long nwin = (long)NPAD_IN * D_MODEL;
    convert_transpose_kernel<<<(unsigned)((nwin + 255) / 256), 256, 0, stream>>>(
        W_in, wtin, D_MODEL, D_IN_PROJ, NPAD_IN);
    long nwout = (long)NPAD_OUT * D_INNER;
    convert_transpose_kernel<<<(unsigned)((nwout + 255) / 256), 256, 0, stream>>>(
        W_out, wtout, D_INNER, D_MODEL, NPAD_OUT);

    // 1) zxbcdt = x @ W_in                (WMMA bf16, TDM-staged B)
    wmma_gemm_bf16<<<dim3((D_IN_PROJ + BN - 1) / BN, ROWS / 64), 128, 0, stream>>>(
        x, wtin, zxbcdt, ROWS, D_IN_PROJ, D_MODEL, NPAD_IN);

    // 2) depthwise conv + SiLU
    long nconv = (long)ROWS * CONV_DIM;
    conv_silu_kernel<<<(unsigned)((nconv + 255) / 256), 256, 0, stream>>>(
        zxbcdt, conv_w, conv_b, xBC);

    // 3) dt = softplus(.)
    dt_kernel<<<(ROWS * NHEADS + 255) / 256, 256, 0, stream>>>(zxbcdt, dt_bias, dtb);

    // 4) selective scan
    scan_kernel<<<BATCH * NHEADS, 256, 0, stream>>>(xBC, dtb, A_log, Dskip, y);

    // 5) gate + RMSNorm (in place on y)
    gate_norm_kernel<<<ROWS, 256, 0, stream>>>(y, zxbcdt, norm_w);

    // 6) yi = y @ W_out                   (WMMA bf16, TDM-staged B)
    wmma_gemm_bf16<<<dim3(D_MODEL / BN, ROWS / 64), 128, 0, stream>>>(
        y, wtout, yi, ROWS, D_MODEL, D_INNER, NPAD_OUT);

    // 7) out = yi @ head_w + head_b
    head_kernel<<<ROWS, 256, 0, stream>>>(yi, head_w, head_b, (float*)d_out);
}
